// DFlashAttention_80212809220228
// MI455X (gfx1250) — matrix operand order repaired and checked by reading, not compiled, not run on hardware
//
#include <hip/hip_runtime.h>

// ---------------------------------------------------------------------------
// Shapes (hardcoded from the reference): B=2, S=1024, CTX=1024, T=2048,
// HID=2048, NH=32, NKV=8, HD=64, rep=4.
// ---------------------------------------------------------------------------

typedef __attribute__((ext_vector_type(16))) __bf16 v16bf;
typedef __attribute__((ext_vector_type(8)))  float  v8f;
typedef __attribute__((ext_vector_type(4)))  unsigned int v4u;
typedef __attribute__((ext_vector_type(4)))  int v4i;

#define AS1 __attribute__((address_space(1)))
#define AS3 __attribute__((address_space(3)))

#if defined(__has_builtin)
#if __has_builtin(__builtin_amdgcn_global_load_async_to_lds_b128) && \
    __has_builtin(__builtin_amdgcn_s_wait_asynccnt)
#define USE_ASYNC_LDS 1
#endif
#endif
#ifndef USE_ASYNC_LDS
#define USE_ASYNC_LDS 0
#endif

struct ABu {            // 32-byte WMMA operand fragment (16 bf16)
  union { v16bf v; v4u q[2]; };
};

__device__ __forceinline__ unsigned short f2bf(float f) {
  unsigned int u = __builtin_bit_cast(unsigned int, f);
  u += 0x7FFFu + ((u >> 16) & 1u);       // round-to-nearest-even
  return (unsigned short)(u >> 16);
}

__device__ __forceinline__ float rowmax16(float x) {
  x = fmaxf(x, __shfl_xor(x, 1, 16));
  x = fmaxf(x, __shfl_xor(x, 2, 16));
  x = fmaxf(x, __shfl_xor(x, 4, 16));
  x = fmaxf(x, __shfl_xor(x, 8, 16));
  return x;
}
__device__ __forceinline__ float rowsum16(float x) {
  x += __shfl_xor(x, 1, 16);
  x += __shfl_xor(x, 2, 16);
  x += __shfl_xor(x, 4, 16);
  x += __shfl_xor(x, 8, 16);
  return x;
}

// 16-byte global -> LDS tile copy: async DMA if available, else sync via regs.
__device__ __forceinline__ void tile_cp16(const unsigned short* src, unsigned short* dst) {
#if USE_ASYNC_LDS
  __builtin_amdgcn_global_load_async_to_lds_b128(
      (AS1 v4i*)src, (AS3 v4i*)dst, 0, 0);
#else
  *(v4u*)dst = *(const v4u*)src;
#endif
}

// ---------------------------------------------------------------------------
// 1) f32 -> bf16 convert (weights)
// ---------------------------------------------------------------------------
__global__ __launch_bounds__(256) void cvt_bf16_kernel(const float* __restrict__ src,
                                                       unsigned short* __restrict__ dst,
                                                       int n) {
  int i = blockIdx.x * 256 + threadIdx.x;
  if (i < n) dst[i] = f2bf(src[i]);
}

// ---------------------------------------------------------------------------
// 2) Build packed bf16 KV activation buffer [B*T, 2048]
// ---------------------------------------------------------------------------
__global__ __launch_bounds__(256) void pack_kv_kernel(const float* __restrict__ hidden,
                                                      const float* __restrict__ ctx,
                                                      unsigned short* __restrict__ kvb) {
  int idx = blockIdx.x * 256 + threadIdx.x;            // 2*2048*2048 total
  int c = idx & 2047;
  int t = (idx >> 11) & 2047;
  int b = idx >> 22;
  float v;
  if (t < 1024) v = ctx[((size_t)(b * 1024 + t)) * 2048 + c];
  else          v = hidden[((size_t)(b * 1024 + (t - 1024))) * 2048 + c];
  kvb[idx] = f2bf(v);
}

// ---------------------------------------------------------------------------
// 3) bf16 GEMM: C[M,N](f32) = A[M,K](bf16) * W[N,K](bf16)^T
//    Block = 256 threads = 8 waves (wave tile 16x64, block tile 128x64).
//    The 64-col W tile is shared by all 8 waves: staged in LDS, async DMA
//    double-buffered; A fragment is register double-buffered.
//    Grids must tile M,N exactly (block-wide barriers inside).
// ---------------------------------------------------------------------------
__global__ __launch_bounds__(256) void gemm_bf16_kernel(const unsigned short* __restrict__ A,
                                                        const unsigned short* __restrict__ W,
                                                        float* __restrict__ C,
                                                        int M, int N, int K, int amode) {
  __shared__ __attribute__((aligned(16))) unsigned short Wlds[2][64][32];

  const int tid  = threadIdx.x;
  const int lane = tid & 31;
  const int wid  = tid >> 5;
  const int ln   = lane & 15;
  const int hi   = lane >> 4;

  const int m0 = (blockIdx.x * 8 + wid) * 16;
  const int n0 = blockIdx.y * 64;

  int m = m0 + ln;
  int arow = (amode == 1) ? ((m >> 10) * 2048 + 1024 + (m & 1023)) : m;
  const unsigned short* aptr = A + (size_t)arow * K + hi * 8;

  // cooperative W-tile staging: thread tid -> col tid/4, 16B quarter tid%4
  const int wc = tid >> 2;
  const int wq = (tid & 3) * 8;
  const unsigned short* wsrc = W + (size_t)(n0 + wc) * K + wq;

  v8f acc[4];
#pragma unroll
  for (int g = 0; g < 4; ++g) acc[g] = (v8f){0.f,0.f,0.f,0.f,0.f,0.f,0.f,0.f};

  ABu a_cur, a_nxt;
  a_cur.q[0] = *(const v4u*)(aptr);
  a_cur.q[1] = *(const v4u*)(aptr + 16);
  a_nxt = a_cur;

  tile_cp16(wsrc, &Wlds[0][wc][wq]);        // chunk 0 -> buf 0

  int cur = 0;
  for (int kb = 0; kb < K; kb += 32) {
    const bool has_next = (kb + 32) < K;
    if (has_next) {
      tile_cp16(wsrc + kb + 32, &Wlds[cur ^ 1][wc][wq]);
      a_nxt.q[0] = *(const v4u*)(aptr + kb + 32);
      a_nxt.q[1] = *(const v4u*)(aptr + kb + 48);
      __builtin_prefetch(aptr + kb + 160, 0, 0);      // global_prefetch_b8
    }
#if USE_ASYNC_LDS
    if (has_next) __builtin_amdgcn_s_wait_asynccnt(1);  // older chunk done
    else          __builtin_amdgcn_s_wait_asynccnt(0);
#endif
    __syncthreads();                         // chunk kb visible to all waves

#pragma unroll
    for (int g = 0; g < 4; ++g) {
      const unsigned short* bp = &Wlds[cur][g * 16 + ln][hi * 8];
      ABu bb;
      bb.q[0] = *(const v4u*)(bp);
      bb.q[1] = *(const v4u*)(bp + 16);
      acc[g] = __builtin_amdgcn_wmma_f32_16x16x32_bf16(
          false, a_cur.v, false, bb.v, (short)0, acc[g], false, false);
    }

    __syncthreads();                         // buf may be overwritten next iter
    a_cur = a_nxt;
    cur ^= 1;
  }

#pragma unroll
  for (int g = 0; g < 4; ++g) {
#pragma unroll
    for (int r = 0; r < 8; ++r) {
      int row = m0 + r + hi * 8;
      int col = n0 + g * 16 + ln;
      C[(size_t)row * N + col] = acc[g][r];
    }
  }
}

// ---------------------------------------------------------------------------
// 4) RMS-norm + RoPE, f32 in -> bf16 out head-major. One wave per head row.
// ---------------------------------------------------------------------------
__global__ __launch_bounds__(256) void norm_rope_kernel(const float* __restrict__ X,
                                                        const float* __restrict__ gamma,
                                                        const float* __restrict__ cosp,
                                                        const float* __restrict__ sinp,
                                                        unsigned short* __restrict__ O,
                                                        int nheads, int seq, int pos_off) {
  const int lane = threadIdx.x & 31;
  const int wid  = threadIdx.x >> 5;
  int gr = blockIdx.x * 8 + wid;                 // (b*seq + s)*nheads + h
  int h = gr % nheads;
  int s = (gr / nheads) % seq;
  int b = gr / (nheads * seq);

  const float* xr = X + ((size_t)(b * seq + s) * nheads + h) * 64;
  float x0 = xr[lane];
  float x1 = xr[lane + 32];

  float ss = x0 * x0 + x1 * x1;
  ss += __shfl_xor(ss, 16, 32);
  ss += __shfl_xor(ss, 8, 32);
  ss += __shfl_xor(ss, 4, 32);
  ss += __shfl_xor(ss, 2, 32);
  ss += __shfl_xor(ss, 1, 32);
  float rs = rsqrtf(ss * (1.0f / 64.0f) + 1e-6f);

  float xn0 = x0 * rs * gamma[lane];
  float xn1 = x1 * rs * gamma[lane + 32];

  int pos = pos_off + s;
  const float* cr = cosp + ((size_t)b * 2048 + pos) * 64;
  const float* sr = sinp + ((size_t)b * 2048 + pos) * 64;
  float c0 = cr[lane], c1 = cr[lane + 32];
  float s0 = sr[lane], s1 = sr[lane + 32];

  float y0 = xn0 * c0 - xn1 * s0;   // rotate_half
  float y1 = xn1 * c1 + xn0 * s1;

  unsigned short* orow = O + ((size_t)(b * nheads + h) * seq + s) * 64;
  orow[lane]      = f2bf(y0);
  orow[lane + 32] = f2bf(y1);
}

// ---------------------------------------------------------------------------
// 5) V transpose + convert: Vt[b][kvh][d][t] bf16
// ---------------------------------------------------------------------------
__global__ __launch_bounds__(256) void vtrans_kernel(const float* __restrict__ Vf,
                                                     unsigned short* __restrict__ Vt) {
  int idx = blockIdx.x * 256 + threadIdx.x;            // 2*8*64*2048 total
  int t = idx & 2047;
  int d = (idx >> 11) & 63;
  int h = (idx >> 17) & 7;
  int b = idx >> 20;
  Vt[idx] = f2bf(Vf[((size_t)(b * 2048 + t)) * 512 + h * 64 + d]);
}

// ---------------------------------------------------------------------------
// 6) Flash attention (GQA). Block = 8 waves, all on the SAME (b, head):
//    8 consecutive 16-row q-tiles share one K/V stream, so the 32-key K and
//    V tiles are staged once per block in LDS (async DMA, double-buffered).
//    Qb [B,32,S,64] ; Kb [B,8,T,64] ; Vt [B,8,64,T] (all bf16)
//    Output: attn bf16 row-major [B*S, 32*64].
// ---------------------------------------------------------------------------
__global__ __launch_bounds__(256) void flash_attn_kernel(const unsigned short* __restrict__ Qb,
                                                         const unsigned short* __restrict__ Kb,
                                                         const unsigned short* __restrict__ Vt,
                                                         unsigned short* __restrict__ Ab) {
  __shared__ __attribute__((aligned(16))) unsigned short Klds[2][32][64]; // [key][dim]
  __shared__ __attribute__((aligned(16))) unsigned short Vlds[2][64][32]; // [dim][t]
  __shared__ __attribute__((aligned(16))) unsigned short Plds[8][16][32];

  const int tid  = threadIdx.x;
  const int lane = tid & 31;
  const int wid  = tid >> 5;
  const int ln   = lane & 15;
  const int hi   = lane >> 4;

  int gw = blockIdx.x * 8 + wid;     // (b*32 + h)*64 + qt ; wid spans qt
  int qt = gw & 63;
  int h  = (gw >> 6) & 31;
  int b  = gw >> 11;
  int kvh = h >> 2;                  // rep = 4, kvh-major grouping

  const float scale = 0.125f;        // 1/sqrt(64)

  // ---- Q tile as two A fragments (head-dim chunks of 32) ----
  int qrow = qt * 16 + ln;
  const unsigned short* qp =
      Qb + ((size_t)((b * 32 + h) * 1024 + qrow)) * 64 + hi * 8;
  ABu aq0, aq1;
  aq0.q[0] = *(const v4u*)(qp +  0);
  aq0.q[1] = *(const v4u*)(qp + 16);
  aq1.q[0] = *(const v4u*)(qp + 32);
  aq1.q[1] = *(const v4u*)(qp + 48);

  const unsigned short* kbase = Kb + ((size_t)(b * 8 + kvh)) * 2048 * 64;
  const unsigned short* vbase = Vt + ((size_t)(b * 8 + kvh)) * 64 * 2048;

  // cooperative staging maps (256 threads x 16B = 4KB per tile)
  const int kr = tid >> 3, kq = (tid & 7) * 8;   // K: key row, 16B slice of 64 dims
  const int vr = tid >> 2, vq = (tid & 3) * 8;   // V: dim row, 16B slice of 32 keys

  v8f O[4];
#pragma unroll
  for (int g = 0; g < 4; ++g) O[g] = (v8f){0.f,0.f,0.f,0.f,0.f,0.f,0.f,0.f};
  float m8[8], l8[8];
#pragma unroll
  for (int r = 0; r < 8; ++r) { m8[r] = -INFINITY; l8[r] = 0.f; }

  // preload chunk 0
  tile_cp16(kbase + (size_t)kr * 64 + kq, &Klds[0][kr][kq]);
  tile_cp16(vbase + (size_t)vr * 2048 + vq, &Vlds[0][vr][vq]);

  int cur = 0;
  for (int tb = 0; tb < 2048; tb += 32) {
    const bool has_next = (tb + 32) < 2048;
    if (has_next) {
      tile_cp16(kbase + (size_t)(tb + 32 + kr) * 64 + kq, &Klds[cur ^ 1][kr][kq]);
      tile_cp16(vbase + (size_t)vr * 2048 + tb + 32 + vq, &Vlds[cur ^ 1][vr][vq]);
    }
#if USE_ASYNC_LDS
    if (has_next) __builtin_amdgcn_s_wait_asynccnt(2);  // older K+V pair done
    else          __builtin_amdgcn_s_wait_asynccnt(0);
#endif
    __syncthreads();

    // ---- scores for two 16-key subtiles from Klds ----
    v8f s0 = (v8f){0.f,0.f,0.f,0.f,0.f,0.f,0.f,0.f};
    v8f s1 = (v8f){0.f,0.f,0.f,0.f,0.f,0.f,0.f,0.f};
    {
      ABu bk;
      const unsigned short* kp = &Klds[cur][ln][hi * 8];
      bk.q[0] = *(const v4u*)(kp);      bk.q[1] = *(const v4u*)(kp + 16);
      s0 = __builtin_amdgcn_wmma_f32_16x16x32_bf16(false, aq0.v, false, bk.v, (short)0, s0, false, false);
      kp = &Klds[cur][ln][32 + hi * 8];
      bk.q[0] = *(const v4u*)(kp);      bk.q[1] = *(const v4u*)(kp + 16);
      s0 = __builtin_amdgcn_wmma_f32_16x16x32_bf16(false, aq1.v, false, bk.v, (short)0, s0, false, false);
      kp = &Klds[cur][16 + ln][hi * 8];
      bk.q[0] = *(const v4u*)(kp);      bk.q[1] = *(const v4u*)(kp + 16);
      s1 = __builtin_amdgcn_wmma_f32_16x16x32_bf16(false, aq0.v, false, bk.v, (short)0, s1, false, false);
      kp = &Klds[cur][16 + ln][32 + hi * 8];
      bk.q[0] = *(const v4u*)(kp);      bk.q[1] = *(const v4u*)(kp + 16);
      s1 = __builtin_amdgcn_wmma_f32_16x16x32_bf16(false, aq1.v, false, bk.v, (short)0, s1, false, false);
    }

    // ---- online softmax update ----
    float corr[8];
#pragma unroll
    for (int r = 0; r < 8; ++r) {
      float a = s0[r] * scale;
      float c = s1[r] * scale;
      s0[r] = a; s1[r] = c;
      float rm = fmaxf(rowmax16(a), rowmax16(c));
      float mn = fmaxf(m8[r], rm);
      corr[r] = __expf(m8[r] - mn);
      m8[r] = mn;
    }
#pragma unroll
    for (int r = 0; r < 8; ++r) {
      float p0 = __expf(s0[r] - m8[r]);
      float p1 = __expf(s1[r] - m8[r]);
      l8[r] = l8[r] * corr[r] + rowsum16(p0) + rowsum16(p1);
      int prow = r + hi * 8;
      Plds[wid][prow][ln]      = f2bf(p0);
      Plds[wid][prow][16 + ln] = f2bf(p1);
    }
#pragma unroll
    for (int g = 0; g < 4; ++g) {
#pragma unroll
      for (int r = 0; r < 8; ++r) O[g][r] *= corr[r];
    }

    asm volatile("s_wait_dscnt 0x0" ::: "memory");   // wave-internal P turnaround

    // ---- P tile back as A fragment (16 x 32) ----
    ABu ap;
    {
      const unsigned short* pp = &Plds[wid][ln][hi * 8];
      ap.q[0] = *(const v4u*)(pp);
      ap.q[1] = *(const v4u*)(pp + 16);
    }

    // ---- PV from Vlds: 4 column groups ----
#pragma unroll
    for (int g = 0; g < 4; ++g) {
      const unsigned short* vp = &Vlds[cur][g * 16 + ln][hi * 8];
      ABu bv;
      bv.q[0] = *(const v4u*)(vp);
      bv.q[1] = *(const v4u*)(vp + 16);
      O[g] = __builtin_amdgcn_wmma_f32_16x16x32_bf16(
          false, ap.v, false, bv.v, (short)0, O[g], false, false);
    }

    __syncthreads();                 // protect double buffer
    cur ^= 1;
  }

  // ---- normalize and emit bf16 attn rows ----
#pragma unroll
  for (int r = 0; r < 8; ++r) {
    float inv = 1.0f / l8[r];
    int row = qt * 16 + r + hi * 8;
#pragma unroll
    for (int g = 0; g < 4; ++g) {
      int col = h * 64 + g * 16 + ln;
      Ab[((size_t)(b * 1024 + row)) * 2048 + col] = f2bf(O[g][r] * inv);
    }
  }
}

// ---------------------------------------------------------------------------
// Host launcher
// ---------------------------------------------------------------------------
extern "C" void kernel_launch(void* const* d_in, const int* in_sizes, int n_in,
                              void* d_out, int out_size, void* d_ws, size_t ws_size,
                              hipStream_t stream) {
  (void)in_sizes; (void)n_in; (void)out_size; (void)ws_size;

  const float* hidden = (const float*)d_in[0];   // [2,1024,2048]
  const float* ctx    = (const float*)d_in[1];   // [2,1024,2048]
  const float* cosp   = (const float*)d_in[2];   // [2,2048,64]
  const float* sinp   = (const float*)d_in[3];   // [2,2048,64]
  const float* Wq     = (const float*)d_in[4];   // [2048,2048]
  const float* Wk     = (const float*)d_in[5];   // [512,2048]
  const float* Wv     = (const float*)d_in[6];   // [512,2048]
  const float* Wo     = (const float*)d_in[7];   // [2048,2048]
  const float* qg     = (const float*)d_in[8];   // [64]
  const float* kg     = (const float*)d_in[9];   // [64]
  float* out = (float*)d_out;                    // [2,1024,2048]

  char* ws = (char*)d_ws;
  size_t off = 0;
  auto take = [&](size_t bytes) -> void* {
    off = (off + 255) & ~(size_t)255;
    void* p = ws + off;
    off += bytes;
    return p;
  };

  unsigned short* KVb = (unsigned short*)take((size_t)2 * 2048 * 2048 * 2);
  unsigned short* Wqb = (unsigned short*)take((size_t)2048 * 2048 * 2);
  unsigned short* Wkb = (unsigned short*)take((size_t)512 * 2048 * 2);
  unsigned short* Wvb = (unsigned short*)take((size_t)512 * 2048 * 2);
  unsigned short* Wob = (unsigned short*)take((size_t)2048 * 2048 * 2);
  float*          Qf  = (float*)take((size_t)2048 * 2048 * 4);
  float*          Kf  = (float*)take((size_t)4096 * 512 * 4);
  float*          Vf  = (float*)take((size_t)4096 * 512 * 4);
  unsigned short* Qb  = (unsigned short*)take((size_t)2 * 32 * 1024 * 64 * 2);
  unsigned short* Kbp = (unsigned short*)take((size_t)2 * 8 * 2048 * 64 * 2);
  unsigned short* Vt  = (unsigned short*)take((size_t)2 * 8 * 64 * 2048 * 2);
  unsigned short* Ab  = (unsigned short*)take((size_t)2048 * 2048 * 2);

  // 1) pack activations + convert weights to bf16
  pack_kv_kernel<<<32768, 256, 0, stream>>>(hidden, ctx, KVb);
  cvt_bf16_kernel<<<16384, 256, 0, stream>>>(Wq, Wqb, 2048 * 2048);
  cvt_bf16_kernel<<<4096, 256, 0, stream>>>(Wk, Wkb, 512 * 2048);
  cvt_bf16_kernel<<<4096, 256, 0, stream>>>(Wv, Wvb, 512 * 2048);
  cvt_bf16_kernel<<<16384, 256, 0, stream>>>(Wo, Wob, 2048 * 2048);

  // 2) projection GEMMs (WMMA bf16 -> f32, LDS-staged W tiles)
  gemm_bf16_kernel<<<dim3(16, 32), 256, 0, stream>>>(KVb, Wqb, Qf, 2048, 2048, 2048, 1);
  gemm_bf16_kernel<<<dim3(32, 8),  256, 0, stream>>>(KVb, Wkb, Kf, 4096, 512, 2048, 0);
  gemm_bf16_kernel<<<dim3(32, 8),  256, 0, stream>>>(KVb, Wvb, Vf, 4096, 512, 2048, 0);

  // 3) RMS-norm + RoPE -> head-major bf16 ; V -> transposed bf16
  norm_rope_kernel<<<8192, 256, 0, stream>>>(Qf, qg, cosp, sinp, Qb, 32, 1024, 1024);
  norm_rope_kernel<<<4096, 256, 0, stream>>>(Kf, kg, cosp, sinp, Kbp, 8, 2048, 0);
  vtrans_kernel<<<8192, 256, 0, stream>>>(Vf, Vt);

  // 4) flash attention (WMMA QK^T + PV, LDS-staged K/V, online softmax)
  flash_attn_kernel<<<512, 256, 0, stream>>>(Qb, Kbp, Vt, Ab);

  // 5) output projection -> f32
  gemm_bf16_kernel<<<dim3(16, 32), 256, 0, stream>>>(Ab, Wob, out, 2048, 2048, 2048, 0);
}
